// RePN_31464930410580
// MI455X (gfx1250) — matrix operand order repaired and checked by reading, not compiled, not run on hardware
//
#include <hip/hip_runtime.h>

#define NDIM 2048
#define PDIM 1024
#define CDIM 91
#define TOPK 128
#define CAND_CAP 16384
#define NBINS 4096

typedef __attribute__((ext_vector_type(16))) __bf16 v16bf;
typedef __attribute__((ext_vector_type(8)))  __bf16 v8bf;
typedef __attribute__((ext_vector_type(8)))  float  v8f;

static __device__ __forceinline__ unsigned short f2bf(float f) {
  unsigned int u = __float_as_uint(f);
  u = u + 0x7FFFu + ((u >> 16) & 1u);   // round-to-nearest-even
  return (unsigned short)(u >> 16);
}

static __device__ __forceinline__ unsigned int f2key(float f) {
  unsigned int u = __float_as_uint(f);
  return (u & 0x80000000u) ? ~u : (u | 0x80000000u);  // monotone total order
}

static __device__ __forceinline__ float iou_box(const float* a, const float* b) {
  float a1 = fmaxf(a[2] - a[0], 0.f) * fmaxf(a[3] - a[1], 0.f);
  float a2 = fmaxf(b[2] - b[0], 0.f) * fmaxf(b[3] - b[1], 0.f);
  float lx = fmaxf(a[0], b[0]), ly = fmaxf(a[1], b[1]);
  float rx = fminf(a[2], b[2]), ry = fminf(a[3], b[3]);
  float w = fmaxf(rx - lx, 0.f), h = fmaxf(ry - ly, 0.f);
  float inter = w * h;
  return inter / (a1 + a2 - inter + 1e-9f);
}

// ---------------- projection: s/o = (scores[:, :90] @ W.T + b) * features, bf16 out
__global__ __launch_bounds__(256) void repn_proj(const float* __restrict__ scores,
                                                 const float* __restrict__ features,
                                                 const float* __restrict__ Wsub,
                                                 const float* __restrict__ bsub,
                                                 const float* __restrict__ Wobj,
                                                 const float* __restrict__ bobj,
                                                 unsigned short* __restrict__ sOut,
                                                 unsigned short* __restrict__ oOut) {
  __shared__ float sc[90];
  const int n = blockIdx.x;
  const int t = threadIdx.x;
  if (t < 90) sc[t] = scores[n * CDIM + t];
  __syncthreads();
#pragma unroll
  for (int q = 0; q < 4; ++q) {
    const int p = t + q * 256;
    float as = bsub[p], ao = bobj[p];
    const float* ws = Wsub + p * 90;
    const float* wo = Wobj + p * 90;
#pragma unroll 6
    for (int c = 0; c < 90; ++c) {
      float v = sc[c];
      as = fmaf(v, ws[c], as);
      ao = fmaf(v, wo[c], ao);
    }
    const float f = features[n * PDIM + p];
    sOut[n * PDIM + p] = f2bf(as * f);
    oOut[n * PDIM + p] = f2bf(ao * f);
  }
}

// ---------------- raw = s @ o^T via v_wmma_f32_16x16x32_bf16, 32x64 tile per wave
__global__ __launch_bounds__(256) void repn_gemm(const unsigned short* __restrict__ sBuf,
                                                 const unsigned short* __restrict__ oBuf,
                                                 float* __restrict__ raw) {
  // block tile: 64 rows x 256 cols; 8 waves arranged 2 (M) x 4 (N)
  if (blockIdx.y * 64 >= blockIdx.x * 256 + 256) return;  // fully lower-triangle: never read
  const int tid = threadIdx.x;
  const int wave = tid >> 5;
  const int lane = tid & 31;
  const int waveM = wave & 1, waveN = wave >> 1;
  const int rowBase = blockIdx.y * 64 + waveM * 32;
  const int colBase = blockIdx.x * 256 + waveN * 64;
  const int laneM = lane & 15;
  const int laneHalf = lane >> 4;

  const __bf16* sp = (const __bf16*)sBuf;
  const __bf16* op = (const __bf16*)oBuf;

  v8f acc[2][4];
#pragma unroll
  for (int mi = 0; mi < 2; ++mi)
#pragma unroll
    for (int ni = 0; ni < 4; ++ni)
      acc[mi][ni] = (v8f){0.f, 0.f, 0.f, 0.f, 0.f, 0.f, 0.f, 0.f};

  for (int kk = 0; kk < PDIM; kk += 32) {
    // A fragment: 16x32, lane L<16 holds row L, K {0..7,16..23}; L>=16 row L-16, K {8..15,24..31}
    v16bf afr[2];
#pragma unroll
    for (int mi = 0; mi < 2; ++mi) {
      const __bf16* base = sp + (size_t)(rowBase + mi * 16 + laneM) * PDIM + kk + laneHalf * 8;
      v8bf c0 = *(const v8bf*)base;
      v8bf c1 = *(const v8bf*)(base + 16);
      afr[mi] = __builtin_shufflevector(c0, c1, 0, 1, 2, 3, 4, 5, 6, 7, 8, 9, 10, 11, 12, 13, 14, 15);
    }
    // B fragment: 32x16 (K x N); lane L<16 holds col L, K {0..7,16..23}; L>=16 col L-16, K {8..15,24..31}
    v16bf bfr[4];
#pragma unroll
    for (int ni = 0; ni < 4; ++ni) {
      const __bf16* base = op + (size_t)(colBase + ni * 16 + laneM) * PDIM + kk + laneHalf * 8;
      v8bf c0 = *(const v8bf*)base;
      v8bf c1 = *(const v8bf*)(base + 16);
      bfr[ni] = __builtin_shufflevector(c0, c1, 0, 1, 2, 3, 4, 5, 6, 7, 8, 9, 10, 11, 12, 13, 14, 15);
    }
#pragma unroll
    for (int mi = 0; mi < 2; ++mi)
#pragma unroll
      for (int ni = 0; ni < 4; ++ni)
        acc[mi][ni] = __builtin_amdgcn_wmma_f32_16x16x32_bf16(
            false, afr[mi], false, bfr[ni], (short)0, acc[mi][ni], false, false);
  }

  // C layout: lanes 0-15: vgpr g -> (M=g, N=lane); lanes 16-31: vgpr g -> (M=8+g, N=lane-16)
#pragma unroll
  for (int mi = 0; mi < 2; ++mi)
#pragma unroll
    for (int ni = 0; ni < 4; ++ni) {
      const int r0 = rowBase + mi * 16 + laneHalf * 8;
      const int c = colBase + ni * 16 + laneM;
#pragma unroll
      for (int g = 0; g < 8; ++g)
        raw[(size_t)(r0 + g) * NDIM + c] = acc[mi][ni][g];
    }
}

// ---------------- top-k machinery
__global__ void repn_init(unsigned int* hist, unsigned int* candCount) {
  for (int i = threadIdx.x; i < NBINS; i += blockDim.x) hist[i] = 0u;
  if (threadIdx.x == 0) *candCount = 0u;
}

__global__ __launch_bounds__(256) void repn_hist(const float* __restrict__ raw,
                                                 unsigned int* __restrict__ hist) {
  __shared__ unsigned int lh[NBINS];
  for (int i = threadIdx.x; i < NBINS; i += blockDim.x) lh[i] = 0u;
  __syncthreads();
  const size_t total = (size_t)NDIM * NDIM;
  for (size_t t = (size_t)blockIdx.x * blockDim.x + threadIdx.x; t < total;
       t += (size_t)gridDim.x * blockDim.x) {
    const int i = (int)(t >> 11);
    const int j = (int)(t & (NDIM - 1));
    if (j > i) atomicAdd(&lh[f2key(raw[t]) >> 20], 1u);
  }
  __syncthreads();
  for (int i = threadIdx.x; i < NBINS; i += blockDim.x)
    if (lh[i]) atomicAdd(&hist[i], lh[i]);
}

__global__ void repn_thresh(const unsigned int* __restrict__ hist, unsigned int* thrKey) {
  if (threadIdx.x == 0 && blockIdx.x == 0) {
    unsigned int cum = 0;
    int bin = NBINS - 1;
    for (; bin >= 0; --bin) {
      cum += hist[bin];
      if (cum >= TOPK) break;
    }
    if (bin < 0) bin = 0;
    *thrKey = (unsigned int)bin << 20;
  }
}

__global__ __launch_bounds__(256) void repn_compact(const float* __restrict__ raw,
                                                    const unsigned int* __restrict__ thrKey,
                                                    unsigned long long* __restrict__ cand,
                                                    unsigned int* __restrict__ candCount) {
  const unsigned int thr = *thrKey;
  const size_t total = (size_t)NDIM * NDIM;
  for (size_t t = (size_t)blockIdx.x * blockDim.x + threadIdx.x; t < total;
       t += (size_t)gridDim.x * blockDim.x) {
    const int i = (int)(t >> 11);
    const int j = (int)(t & (NDIM - 1));
    if (j > i) {
      const unsigned int key = f2key(raw[t]);
      if (key >= thr) {
        const unsigned int pos = atomicAdd(candCount, 1u);
        if (pos < CAND_CAP)
          cand[pos] = ((unsigned long long)key << 32) | (0xFFFFFFFFu - (unsigned int)t);
      }
    }
  }
}

__global__ __launch_bounds__(256) void repn_select(unsigned long long* __restrict__ cand,
                                                   const unsigned int* __restrict__ candCount,
                                                   int* __restrict__ pairs) {
  __shared__ unsigned long long sval[256];
  __shared__ int spos[256];
  const int t = threadIdx.x;
  unsigned int ncU = *candCount;
  const int nc = (int)(ncU < (unsigned)CAND_CAP ? ncU : (unsigned)CAND_CAP);
  for (int k = 0; k < TOPK; ++k) {
    unsigned long long best = 0ULL;
    int bp = -1;
    for (int i = t; i < nc; i += 256) {
      unsigned long long v = cand[i];
      if (v > best) { best = v; bp = i; }
    }
    sval[t] = best;
    spos[t] = bp;
    __syncthreads();
    for (int s = 128; s > 0; s >>= 1) {
      if (t < s && sval[t + s] > sval[t]) { sval[t] = sval[t + s]; spos[t] = spos[t + s]; }
      __syncthreads();
    }
    if (t == 0) {
      unsigned long long b = sval[0];
      int p = spos[0];
      pairs[k] = (b == 0ULL || p < 0) ? 0 : (int)(0xFFFFFFFFu - (unsigned int)(b & 0xFFFFFFFFULL));
      if (p >= 0) cand[p] = 0ULL;
    }
    __syncthreads();
  }
}

// ---------------- greedy pair NMS + "larger box" selection
__global__ __launch_bounds__(128) void repn_nms(const float* __restrict__ boxes,
                                                const int* __restrict__ pairs,
                                                int* __restrict__ keep,
                                                int* __restrict__ selrow) {
  __shared__ float sb[TOPK][4], ob[TOPK][4];
  __shared__ int kflag[TOPK];
  const int t = threadIdx.x;
  const int idx = pairs[t];
  const int subj = idx >> 11;
  const int obj = idx & (NDIM - 1);
#pragma unroll
  for (int c = 0; c < 4; ++c) {
    sb[t][c] = boxes[subj * 4 + c];
    ob[t][c] = boxes[obj * 4 + c];
  }
  kflag[t] = 1;
  __syncthreads();
  for (int i = 0; i < TOPK; ++i) {
    if (t > i && kflag[t] && kflag[i]) {
      if (iou_box(sb[i], sb[t]) > 0.7f && iou_box(ob[i], ob[t]) > 0.7f) kflag[t] = 0;
    }
    __syncthreads();
  }
  keep[t] = kflag[t];
  float ps = sb[t][0] - sb[t][1]; ps *= ps;   // |x1-y1|^2 proxy
  float po = ob[t][0] - ob[t][1]; po *= po;
  selrow[t] = (ps >= po) ? subj : obj;        // argmax ties -> subj
}

// ---------------- masked gathers to outputs
__global__ __launch_bounds__(256) void repn_gather(const float* __restrict__ boxes,
                                                   const float* __restrict__ scores,
                                                   const float* __restrict__ features,
                                                   const int* __restrict__ pairs,
                                                   const int* __restrict__ keep,
                                                   const int* __restrict__ selrow,
                                                   float* __restrict__ outBoxes,
                                                   float* __restrict__ outFeats,
                                                   float* __restrict__ outScores) {
  const int k = blockIdx.x;
  const int t = threadIdx.x;
  const int idx = pairs[k];
  const int subj = idx >> 11;
  const int obj = idx & (NDIM - 1);
  const float m = keep[k] ? 1.0f : 0.0f;
  const int sel = selrow[k];
  if (t < 8) {
    const int half = t >> 2, c = t & 3;
    const int bi = half ? obj : subj;
    outBoxes[k * 8 + t] = boxes[bi * 4 + c] * m;
  }
  if (t < CDIM) outScores[k * CDIM + t] = scores[sel * CDIM + t] * m;
#pragma unroll
  for (int q = 0; q < 4; ++q) {
    const int p = t + q * 256;
    outFeats[k * PDIM + p] = features[sel * PDIM + p] * m;
  }
}

extern "C" void kernel_launch(void* const* d_in, const int* in_sizes, int n_in,
                              void* d_out, int out_size, void* d_ws, size_t ws_size,
                              hipStream_t stream) {
  (void)in_sizes; (void)n_in; (void)out_size; (void)ws_size;
  const float* boxes    = (const float*)d_in[0];  // [8,2048,4]
  const float* scores   = (const float*)d_in[1];  // [8,2048,91]
  const float* features = (const float*)d_in[2];  // [8,2048,1024]
  const float* Wsub     = (const float*)d_in[3];  // [1024,90]
  const float* bsub     = (const float*)d_in[4];  // [1024]
  const float* Wobj     = (const float*)d_in[5];  // [1024,90]
  const float* bobj     = (const float*)d_in[6];  // [1024]

  float* outBoxes  = (float*)d_out;                              // [8,128,2,4]
  float* outFeats  = outBoxes + 8 * TOPK * 8;                    // [8,128,1024]
  float* outScores = outFeats + 8 * TOPK * PDIM;                 // [8,128,91]

  // workspace layout (reused across batches; stream-serialized)
  char* w = (char*)d_ws;
  size_t off = 0;
  auto alloc = [&](size_t bytes) { void* p = w + off; off = (off + bytes + 255) & ~(size_t)255; return p; };
  unsigned short* sBuf = (unsigned short*)alloc((size_t)NDIM * PDIM * 2);
  unsigned short* oBuf = (unsigned short*)alloc((size_t)NDIM * PDIM * 2);
  float* raw           = (float*)alloc((size_t)NDIM * NDIM * 4);
  unsigned int* hist   = (unsigned int*)alloc(NBINS * 4);
  unsigned int* thrKey = (unsigned int*)alloc(4);
  unsigned int* candCt = (unsigned int*)alloc(4);
  unsigned long long* cand = (unsigned long long*)alloc((size_t)CAND_CAP * 8);
  int* pairs  = (int*)alloc(TOPK * 4);
  int* keep   = (int*)alloc(TOPK * 4);
  int* selrow = (int*)alloc(TOPK * 4);

  for (int b = 0; b < 8; ++b) {
    const float* bx = boxes    + (size_t)b * NDIM * 4;
    const float* sc = scores   + (size_t)b * NDIM * CDIM;
    const float* ft = features + (size_t)b * NDIM * PDIM;

    repn_proj<<<NDIM, 256, 0, stream>>>(sc, ft, Wsub, bsub, Wobj, bobj, sBuf, oBuf);
    repn_gemm<<<dim3(NDIM / 256, NDIM / 64), 256, 0, stream>>>(sBuf, oBuf, raw);
    repn_init<<<1, 256, 0, stream>>>(hist, candCt);
    repn_hist<<<512, 256, 0, stream>>>(raw, hist);
    repn_thresh<<<1, 32, 0, stream>>>(hist, thrKey);
    repn_compact<<<512, 256, 0, stream>>>(raw, thrKey, cand, candCt);
    repn_select<<<1, 256, 0, stream>>>(cand, candCt, pairs);
    repn_nms<<<1, 128, 0, stream>>>(bx, pairs, keep, selrow);
    repn_gather<<<TOPK, 256, 0, stream>>>(bx, sc, ft, pairs, keep, selrow,
                                          outBoxes + (size_t)b * TOPK * 8,
                                          outFeats + (size_t)b * TOPK * PDIM,
                                          outScores + (size_t)b * TOPK * CDIM);
  }
}
